// ScaledDotProductAttentionPyT_82764019794105
// MI455X (gfx1250) — compile-verified, matrix operand order checked by reading, not run on hardware
//
#include <hip/hip_runtime.h>
#include <math.h>

typedef __attribute__((ext_vector_type(16))) _Float16 v16h;
typedef __attribute__((ext_vector_type(8)))  _Float16 v8h;
typedef __attribute__((ext_vector_type(4)))  _Float16 v4h;
typedef __attribute__((ext_vector_type(8)))  float    v8f;

#define B_ 4
#define H_ 16
#define S_ 2048
#define D_ 128

// ---------------------------------------------------------------------------
// Pass 1: K fp32 -> f16 (same [bh,s,d] layout).
//         V fp32 -> f16 TRANSPOSED [bh,d,s], with the KV index PERMUTED inside
//         each 64-block: storage p <-> logical j = (p>>2) + 16*(p&3).
// The permutation makes the attention kernel's P-tile LDS stores contiguous
// (one b64 per row) while keeping P(A-frag) and V^T(B-frag) k-orders matched.
// ---------------------------------------------------------------------------
__global__ __launch_bounds__(256) void cvt_kv_kernel(const float* __restrict__ k,
                                                     const float* __restrict__ v,
                                                     _Float16* __restrict__ kh,
                                                     _Float16* __restrict__ vt) {
  const long i  = (long)blockIdx.x * blockDim.x + threadIdx.x;
  const long SD = (long)S_ * D_;
  kh[i] = (_Float16)k[i];
  const long bh  = i / SD;
  const long rem = i - bh * SD;
  const long dd  = rem / S_;
  const long ss  = rem - dd * S_;         // storage kv index
  const long p   = ss & 63;
  const long jj  = (ss & ~63L) + (p >> 2) + ((p & 3) << 4);  // logical kv index
  vt[i] = (_Float16)v[bh * SD + jj * D_ + dd];
}

// 16x32 f16 fragment load (A or B operand, CDNA5 wave32 layout). Caller folds
// the per-lane row offset and half*8 chunk offset into p: chunk0 at p[0..7],
// chunk1 at p[16..23].
__device__ inline v16h load_frag(const _Float16* p) {
  v8h lo = *(const v8h*)(p);
  v8h hi = *(const v8h*)(p + 16);
  v16h a;
#pragma unroll
  for (int j = 0; j < 8; ++j) { a[j] = lo[j]; a[8 + j] = hi[j]; }
  return a;
}

// ---------------------------------------------------------------------------
// Pass 2: flash attention. One wave per 16-row Q tile, KV blocked by 64.
// ---------------------------------------------------------------------------
__global__ __launch_bounds__(256) void attn_kernel(const float* __restrict__ q,
                                                   const _Float16* __restrict__ kh,
                                                   const _Float16* __restrict__ vt,
                                                   float* __restrict__ out) {
  __shared__ _Float16 lds[8][16 * 72];   // per-wave 16x64 P tile (+pad), 144B rows

  const int lane = threadIdx.x & 31;
  const int wave = threadIdx.x >> 5;
  const int half = lane >> 4;
  const int lc   = lane & 15;

  const int tile = blockIdx.x * 8 + wave;          // 8192 tiles
  const int bh   = tile >> 7;
  const int q0   = (tile & 127) << 4;

  const float LOG2E  = 1.4426950408889634f;
  const float slope2 = exp2f(-0.5f * (float)((bh & (H_ - 1)) + 1)) * LOG2E;
  const float scale2 = 0.08838834764831845f * LOG2E;   // 1/sqrt(128) * log2(e)

  const long base = (long)bh * S_ * D_;

  // ---- Q fragments: fp32 load, softmax scale folded in, cvt to f16 ----
  v16h qa[4];
  const float* qrow = q + base + (long)(q0 + lc) * D_ + half * 8;
#pragma unroll
  for (int dc = 0; dc < 4; ++dc) {
    v16h a;
#pragma unroll
    for (int j = 0; j < 8; ++j) {
      a[j]     = (_Float16)(qrow[dc * 32 + j]      * scale2);
      a[8 + j] = (_Float16)(qrow[dc * 32 + 16 + j] * scale2);
    }
    qa[dc] = a;
  }

  v8f o[8];
#pragma unroll
  for (int i = 0; i < 8; ++i) o[i] = {};
  float mrow[8], lsum[8], bi[8];
#pragma unroll
  for (int v = 0; v < 8; ++v) {
    mrow[v] = -3.0e38f;
    lsum[v] = 0.0f;                                   // per-lane partial denominator
    bi[v]   = -(float)(q0 + v + 8 * half) * slope2;   // -i * slope (log2 domain)
  }
  float aj[4];
#pragma unroll
  for (int t = 0; t < 4; ++t) aj[t] = (float)(16 * t + lc) * slope2;  // +j * slope
  const float ajinc = 64.0f * slope2;

  // Per-lane base pointers (row + K-chunk offsets folded in once).
  const _Float16* kbase = kh + base + (long)lc * D_ + half * 8;
  const _Float16* vbase = vt + base + (long)lc * S_ + half * 8;
  _Float16*       lw = &lds[wave][(8 * half) * 72 + 4 * lc];  // packed P writer
  const _Float16* lr = &lds[wave][lc * 72 + half * 8];        // A-layout reader

  auto do_block = [&](int kv0, bool masked) {
    const _Float16* kp = kbase + (long)kv0 * D_;
    const _Float16* vp = vbase + kv0;

    // ---- S = Q K^T : four 16x16 C tiles over a 64-wide KV block ----
    v8f sf[4];
#pragma unroll
    for (int t = 0; t < 4; ++t) {
      v8f sc = {};
#pragma unroll
      for (int dc = 0; dc < 4; ++dc) {
        v16h kb = load_frag(kp + 16 * t * D_ + dc * 32);
        sc = __builtin_amdgcn_wmma_f32_16x16x32_f16(false, qa[dc], false, kb,
                                                    (short)0, sc, false, false);
      }
      sf[t] = sc;
    }

    // ---- ALiBi bias + online softmax (exp2 domain); mask only when peeled ----
    float alphav[8];
#pragma unroll
    for (int v = 0; v < 8; ++v) {
      const float b = bi[v];
      float s0 = sf[0][v] + (aj[0] + b);
      float s1 = sf[1][v] + (aj[1] + b);
      float s2 = sf[2][v] + (aj[2] + b);
      float s3 = sf[3][v] + (aj[3] + b);
      if (masked) {
        const int gi = q0 + v + 8 * half;
        s0 = (kv0 + lc      > gi) ? -3.0e38f : s0;
        s1 = (kv0 + 16 + lc > gi) ? -3.0e38f : s1;
        s2 = (kv0 + 32 + lc > gi) ? -3.0e38f : s2;
        s3 = (kv0 + 48 + lc > gi) ? -3.0e38f : s3;
      }
      float x = fmaxf(fmaxf(s0, s1), fmaxf(s2, s3));
#pragma unroll
      for (int off = 1; off < 16; off <<= 1) x = fmaxf(x, __shfl_xor(x, off, 32));
      const float mn = fmaxf(mrow[v], x);       // group-uniform
      const float al = exp2f(mrow[v] - mn);     // group-uniform
      const float e0 = exp2f(s0 - mn);
      const float e1 = exp2f(s1 - mn);
      const float e2 = exp2f(s2 - mn);
      const float e3 = exp2f(s3 - mn);
      // Deferred denominator: per-lane partial, reduced once after the loop.
      lsum[v]   = lsum[v] * al + ((e0 + e1) + (e2 + e3));
      mrow[v]   = mn;
      alphav[v] = al;
      // Packed P store: storage cols 4*lc..4*lc+3 <-> logical cols lc+16t.
      v4h pk = { (_Float16)e0, (_Float16)e1, (_Float16)e2, (_Float16)e3 };
      *(v4h*)(lw + v * 72) = pk;
    }

    // ---- rescale O accumulators ----
#pragma unroll
    for (int d16 = 0; d16 < 8; ++d16)
#pragma unroll
      for (int v = 0; v < 8; ++v) o[d16][v] *= alphav[v];

    asm volatile("s_wait_dscnt 0" ::: "memory");   // wave-local store->load ordering
    v16h pa0 = load_frag(lr + 0);
    v16h pa1 = load_frag(lr + 32);
    asm volatile("" ::: "memory");

    // ---- O += P * V (two permuted-KV chunks per d16 column tile) ----
#pragma unroll
    for (int d16 = 0; d16 < 8; ++d16) {
      v16h vb0 = load_frag(vp + d16 * 16 * S_);
      v16h vb1 = load_frag(vp + d16 * 16 * S_ + 32);
      o[d16] = __builtin_amdgcn_wmma_f32_16x16x32_f16(false, pa0, false, vb0,
                                                      (short)0, o[d16], false, false);
      o[d16] = __builtin_amdgcn_wmma_f32_16x16x32_f16(false, pa1, false, vb1,
                                                      (short)0, o[d16], false, false);
    }

#pragma unroll
    for (int t = 0; t < 4; ++t) aj[t] += ajinc;
  };

  // Unmasked full blocks, then exactly one diagonal (masked) block.
  const int nmain = q0 >> 6;
  for (int b = 0; b < nmain; ++b) do_block(b * 64, false);
  do_block(nmain * 64, true);

  // ---- final denominator reduction, normalize, store fp32 ----
  float rl[8];
#pragma unroll
  for (int v = 0; v < 8; ++v) {
    float s = lsum[v];
#pragma unroll
    for (int off = 1; off < 16; off <<= 1) s += __shfl_xor(s, off, 32);
    rl[v] = 1.0f / s;
  }
  float* orow = out + base + (long)(q0 + 8 * half) * D_ + lc;
#pragma unroll
  for (int d16 = 0; d16 < 8; ++d16) {
#pragma unroll
    for (int v = 0; v < 8; ++v) {
      orow[(long)v * D_ + d16 * 16] = o[d16][v] * rl[v];
    }
  }
}

extern "C" void kernel_launch(void* const* d_in, const int* in_sizes, int n_in,
                              void* d_out, int out_size, void* d_ws, size_t ws_size,
                              hipStream_t stream) {
  const float* q = (const float*)d_in[0];
  const float* k = (const float*)d_in[1];
  const float* v = (const float*)d_in[2];
  float* out = (float*)d_out;

  const long total = (long)B_ * H_ * S_ * D_;      // 16,777,216 elements
  // Workspace: K(f16) 32MB + V^T(f16, kv-permuted) 32MB = 64MB
  _Float16* kh = (_Float16*)d_ws;
  _Float16* vt = kh + total;

  cvt_kv_kernel<<<(int)(total / 256), 256, 0, stream>>>(k, v, kh, vt);

  const int tiles = B_ * H_ * (S_ / 16);           // 8192 Q tiles
  attn_kernel<<<tiles / 8, 256, 0, stream>>>(q, kh, vt, out);
}
